// Attention_16681652977913
// MI455X (gfx1250) — compile-verified, matrix operand order checked by reading, not run on hardware
//
#include <hip/hip_runtime.h>

// ---------------- problem constants ----------------
constexpr int BB   = 2;
constexpr int TT   = 2048;
constexpr int DM   = 2048;   // d_model
constexpr int DH   = 128;    // d_head
constexpr int NQ   = 16;
constexpr int NKV  = 4;
constexpr int MROWS = BB * TT;           // 4096 GEMM rows
#define ATTN_SCALE 0.07762672f           // 1/((0.1*ln4+1)*sqrt(128))
#define RMS_EPS    1.1920929e-07f

// ---------------- WMMA types ----------------
typedef __attribute__((ext_vector_type(16))) __bf16          v16bf;
typedef __attribute__((ext_vector_type(16))) unsigned short  v16u;
typedef __attribute__((ext_vector_type(8)))  float           v8f;

__device__ inline unsigned short f2bf_u(float f) {
  unsigned u = __float_as_uint(f);
  unsigned r = u + 0x7FFFu + ((u >> 16) & 1u);   // round-to-nearest-even
  return (unsigned short)(r >> 16);
}
__device__ inline unsigned pack2(float a, float b) {
  return (unsigned)f2bf_u(a) | ((unsigned)f2bf_u(b) << 16);
}

__device__ inline v8f wmma_bf16(v16bf a, v16bf b, v8f c) {
  return __builtin_amdgcn_wmma_f32_16x16x32_bf16(
      false, a, false, b, (short)0, c, false, false);
}

// ---------------- CDNA5 async global->LDS copy ----------------
__device__ inline unsigned lds_addr_of(const void* p) {
  // generic -> LDS address space; AS(3) pointers are 32-bit LDS offsets
  return (unsigned)(unsigned long long)(__attribute__((address_space(3))) const char*)p;
}
__device__ inline void async_load_b128(unsigned lds_byte_addr, const void* gptr) {
  asm volatile("global_load_async_to_lds_b128 %0, %1, off"
               :: "v"(lds_byte_addr), "v"(gptr)
               : "memory");
}
__device__ inline void wait_asynccnt0() {
  asm volatile("s_wait_asynccnt 0" ::: "memory");
}
__device__ inline void wait_asynccnt4() {
  asm volatile("s_wait_asynccnt 4" ::: "memory");
}

// A fragment: 16x32 bf16 tile, row-major in LDS (stride ldstride shorts).
// ISA 7.12.2: lanes 0-15 hold M=lane, lanes 16-31 M=lane-16;
// VGPR j pair -> K = h*8+2j (+16 for upper 4 VGPRs), h = lane>>4.
// Each half is 8 contiguous shorts -> lowers to 2x ds_load_b128.
__device__ inline v16bf load_frag_a(const unsigned short* lds, int ldstride) {
  int lane = threadIdx.x & 31;
  int m = lane & 15, h = lane >> 4;
  v16u t;
#pragma unroll
  for (int j = 0; j < 4; ++j) {
    int k0 = h * 8 + 2 * j;
    t[2*j]     = lds[m * ldstride + k0];
    t[2*j + 1] = lds[m * ldstride + k0 + 1];
    t[8 + 2*j] = lds[m * ldstride + 16 + k0];
    t[9 + 2*j] = lds[m * ldstride + 16 + k0 + 1];
  }
  return __builtin_bit_cast(v16bf, t);
}

// B fragment: 32x16 bf16 tile, K-major in LDS (row stride ldstride shorts).
__device__ inline v16bf load_frag_b(const unsigned short* lds, int ldstride) {
  int lane = threadIdx.x & 31;
  int n = lane & 15, h = lane >> 4;
  v16u t;
#pragma unroll
  for (int j = 0; j < 4; ++j) {
    int k0 = h * 8 + 2 * j;
    t[2*j]     = lds[(k0)     * ldstride + n];
    t[2*j + 1] = lds[(k0 + 1) * ldstride + n];
    t[8 + 2*j] = lds[(k0 + 16) * ldstride + n];
    t[9 + 2*j] = lds[(k0 + 17) * ldstride + n];
  }
  return __builtin_bit_cast(v16bf, t);
}

// ---------------- generic bf16 WMMA GEMM ----------------
// C(MxN fp32) = A(MxK) * B(KxN fp32->bf16). A fp32 or bf16(u16) per template.
// Block 64x64, 8 waves (2x4), each wave -> 32x16 output (two 16x16 frags).
template <bool A_IS_BF16>
__global__ void __launch_bounds__(256, 1)
gemm_wmma(const void* __restrict__ A_,
          const float* __restrict__ B,
          float* __restrict__ C,
          int M, int N, int K) {
  constexpr int BM = 64, BN = 64, BK = 32;
  __shared__ alignas(16) unsigned short As[BM * BK];  // 64x32
  __shared__ alignas(16) unsigned short Bs[BK * BN];  // 32x64
  const int tid  = threadIdx.x;
  const int wave = tid >> 5;
  const int wm   = wave & 1;    // 0..1
  const int wn   = wave >> 1;   // 0..3
  const int bm   = blockIdx.x * BM;
  const int bn   = blockIdx.y * BN;

  const float*          Af = (const float*)A_;
  const unsigned short* Ab = (const unsigned short*)A_;

  v8f acc0 = {}, acc1 = {};

  for (int kb = 0; kb < K; kb += BK) {
    // stage A tile 64x32 in 4-element chunks (2 chunks/thread)
#pragma unroll
    for (int e = 0; e < 2; ++e) {
      int ci = tid + 256 * e;     // 0..511
      int r  = ci >> 3;           // 8 chunks per 32-wide row
      int c4 = (ci & 7) * 4;
      if constexpr (A_IS_BF16) {
        uint2 v = *(const uint2*)&Ab[(size_t)(bm + r) * K + kb + c4];
        *(uint2*)&As[r * BK + c4] = v;
      } else {
        float4 f = *(const float4*)&Af[(size_t)(bm + r) * K + kb + c4];
        uint2 v; v.x = pack2(f.x, f.y); v.y = pack2(f.z, f.w);
        *(uint2*)&As[r * BK + c4] = v;
      }
    }
    // stage B tile 32x64 fp32 -> bf16
#pragma unroll
    for (int e = 0; e < 2; ++e) {
      int ci = tid + 256 * e;
      int r  = ci >> 4;           // 16 chunks per 64-wide row
      int c4 = (ci & 15) * 4;
      float4 f = *(const float4*)&B[(size_t)(kb + r) * N + bn + c4];
      uint2 v; v.x = pack2(f.x, f.y); v.y = pack2(f.z, f.w);
      *(uint2*)&Bs[r * BN + c4] = v;
    }
    if (kb + BK < K) {   // hint next B tile -> global_prefetch_b8
      __builtin_prefetch(&B[(size_t)(kb + BK) * N + bn], 0, 1);
    }
    __syncthreads();

    v16bf bf = load_frag_b(&Bs[wn * 16], BN);
    v16bf a0 = load_frag_a(&As[(wm * 32) * BK], BK);
    v16bf a1 = load_frag_a(&As[(wm * 32 + 16) * BK], BK);
    acc0 = wmma_bf16(a0, bf, acc0);
    acc1 = wmma_bf16(a1, bf, acc1);
    __syncthreads();
  }

  // C layout: VGPR r -> row (h?r+8:r), col = lane&15 (ISA 7.12.2)
  const int lane = tid & 31;
  const int n  = lane & 15;
  const int hh = lane >> 4;
  const int col = bn + wn * 16 + n;
#pragma unroll
  for (int r = 0; r < 8; ++r) {
    int row0 = bm + wm * 32 + (hh ? r + 8 : r);
    C[(size_t)row0 * N + col]        = acc0[r];
    C[(size_t)(row0 + 16) * N + col] = acc1[r];
  }
}

// ---------------- RMSNorm + YaRN RoPE + bf16 pack ----------------
// One block (128 threads) per (b,t,slot). slot 0-15: q head, 16-19: k group,
// 20-23: v group (cast only). Outputs head-major bf16.
__global__ void __launch_bounds__(128)
normrope(const float* __restrict__ qproj,
         const float* __restrict__ kvproj,
         const float* __restrict__ qw,
         const float* __restrict__ kw,
         unsigned short* __restrict__ qn,
         unsigned short* __restrict__ kn,
         unsigned short* __restrict__ vn) {
  const int row  = blockIdx.x;      // 0 .. BB*TT*24-1
  const int slot = row % 24;
  const int bt   = row / 24;        // b*TT + t
  const int t    = bt % TT;
  const int b    = bt / TT;
  const int d    = threadIdx.x;     // 0..127

  __shared__ float srow[DH];
  __shared__ float partial[4];

  float x;
  const float* w = nullptr;
  if (slot < 16)      { x = qproj[(size_t)bt * (NQ * DH) + slot * DH + d]; w = qw; }
  else if (slot < 20) { x = kvproj[(size_t)bt * (2 * NKV * DH) + (slot - 16) * DH + d]; w = kw; }
  else                { x = kvproj[(size_t)bt * (2 * NKV * DH) + NKV * DH + (slot - 20) * DH + d]; }

  unsigned short outv;
  if (slot < 20) {
    srow[d] = x;
    float ss = x * x;
#pragma unroll
    for (int msk = 16; msk >= 1; msk >>= 1) ss += __shfl_xor(ss, msk, 32);
    if ((d & 31) == 0) partial[d >> 5] = ss;
    __syncthreads();
    float tot = partial[0] + partial[1] + partial[2] + partial[3];
    float inv_rms = rsqrtf(tot * (1.0f / DH) + RMS_EPS);
    float xn = x * inv_rms * w[d];

    int   pd  = (d < 64) ? d + 64 : d - 64;
    float xpn = srow[pd] * inv_rms * w[pd];
    float rot = (d < 64) ? -xpn : xpn;

    // YaRN inv_freq for this dim
    int   idx = d & 63;
    float inv = __powf(10000.0f, -(float)idx * (1.0f / 64.0f));
    float wl  = 6.283185307179586f / inv;
    float gamma = (4096.0f - wl) * (1.0f / (4096.0f - 128.0f));
    gamma = fminf(fmaxf(gamma, 0.0f), 1.0f);
    float invy = gamma * inv + (1.0f - gamma) * inv * 0.25f;
    float f = (float)t * invy;
    float s, c;
    __sincosf(f, &s, &c);
    outv = f2bf_u(xn * c + rot * s);
  } else {
    outv = f2bf_u(x);
  }

  if (slot < 16)
    qn[(((size_t)(b * NQ + slot)) * TT + t) * DH + d] = outv;
  else if (slot < 20)
    kn[(((size_t)(b * NKV + slot - 16)) * TT + t) * DH + d] = outv;
  else
    vn[(((size_t)(b * NKV + slot - 20)) * TT + t) * DH + d] = outv;
}

// ---------------- flash attention (bf16 WMMA, causal, GQA) ----------------
// grid = (TT/128, NQ, BB); 256 threads = 8 waves; wave w owns 16 query rows.
// Q kept in LDS; K/V tiles double-buffered and streamed with CDNA5 async
// global->LDS b128 copies (ASYNCcnt), overlapping DMA with WMMA compute.
__global__ void __launch_bounds__(256, 1)
attn_wmma(const unsigned short* __restrict__ qn,
          const unsigned short* __restrict__ kn,
          const unsigned short* __restrict__ vn,
          unsigned short* __restrict__ attn_out) {
  constexpr int QT = 128, KT = 32;
  const int qtile = blockIdx.x;
  const int h     = blockIdx.y;
  const int b     = blockIdx.z;
  const int g     = h % NKV;            // h = qpg*NKV + g
  const int tid  = threadIdx.x;
  const int lane = tid & 31;
  const int wave = tid >> 5;
  const int m    = lane & 15;
  const int hh   = lane >> 4;

  const unsigned short* Kg = kn + ((size_t)(b * NKV + g)) * TT * DH;
  const unsigned short* Vg = vn + ((size_t)(b * NKV + g)) * TT * DH;

  __shared__ alignas(16) unsigned short Qs[QT * DH];          // 32 KB
  __shared__ alignas(16) unsigned short Ks[2][KT * DH];       // 2 x 8 KB
  __shared__ alignas(16) unsigned short Vs[2][KT * DH];       // 2 x 8 KB
  __shared__ alignas(16) unsigned short Ps[8][16 * 32];       // 8 KB

  // async-stage the whole 128x128 Q tile (32 KB, 16B/thread x 8)
  {
    const char* qg = (const char*)(qn + (((size_t)(b * NQ + h)) * TT + qtile * QT) * DH);
    unsigned ldsQ = lds_addr_of(Qs);
#pragma unroll
    for (int e = 0; e < 8; ++e) {
      int off = (tid + 256 * e) * 16;
      async_load_b128(ldsQ + off, qg + off);
    }
  }
  // prime the pipeline: tile 0 into buffer 0 (4 async issues per wave)
  {
    const char* kg = (const char*)Kg;
    const char* vg = (const char*)Vg;
    unsigned ldsK = lds_addr_of(Ks[0]);
    unsigned ldsV = lds_addr_of(Vs[0]);
#pragma unroll
    for (int e = 0; e < 2; ++e) {
      int off = (tid + 256 * e) * 16;
      async_load_b128(ldsK + off, kg + off);
      async_load_b128(ldsV + off, vg + off);
    }
  }

  v8f o[8];
#pragma unroll
  for (int f = 0; f < 8; ++f) o[f] = v8f{};
  float mstat[8], lstat[8];
#pragma unroll
  for (int r = 0; r < 8; ++r) { mstat[r] = -__builtin_inff(); lstat[r] = 0.0f; }

  const int qrow_base = qtile * QT + wave * 16 + (hh ? 8 : 0);  // row of VGPR r
  const int ntiles = (qtile * QT + QT) / KT;  // block-uniform causal bound

  for (int it = 0; it < ntiles; ++it) {
    const int cur = it & 1;
    // issue next tile's async loads into the other buffer (safe: end-of-iter
    // barrier of it-1 guarantees nobody still reads buffer cur^1)
    if (it + 1 < ntiles) {
      const char* kg = (const char*)(Kg + (size_t)(it + 1) * KT * DH);
      const char* vg = (const char*)(Vg + (size_t)(it + 1) * KT * DH);
      unsigned ldsK = lds_addr_of(Ks[cur ^ 1]);
      unsigned ldsV = lds_addr_of(Vs[cur ^ 1]);
#pragma unroll
      for (int e = 0; e < 2; ++e) {
        int off = (tid + 256 * e) * 16;
        async_load_b128(ldsK + off, kg + off);
        async_load_b128(ldsV + off, vg + off);
      }
      wait_asynccnt4();   // in-order: leaves only the 4 just-issued in flight
    } else {
      wait_asynccnt0();
    }
    __syncthreads();

    const unsigned short* Kt = Ks[cur];
    const unsigned short* Vt = Vs[cur];
    const int kb = it * KT;

    // S = Q * K^T : two 16x16 frags (keys kb..kb+15 and kb+16..kb+31)
    v8f s0 = v8f{}, s1 = v8f{};
#pragma unroll
    for (int ks = 0; ks < 4; ++ks) {
      v16bf qa = load_frag_a(&Qs[(wave * 16) * DH + ks * 32], DH);
      v16u t0, t1;
#pragma unroll
      for (int j = 0; j < 4; ++j) {
        int k0 = ks * 32 + hh * 8 + 2 * j;
        t0[2*j]     = Kt[m * DH + k0];
        t0[2*j + 1] = Kt[m * DH + k0 + 1];
        t0[8 + 2*j] = Kt[m * DH + 16 + k0];
        t0[9 + 2*j] = Kt[m * DH + 17 + k0];
        t1[2*j]     = Kt[(16 + m) * DH + k0];
        t1[2*j + 1] = Kt[(16 + m) * DH + k0 + 1];
        t1[8 + 2*j] = Kt[(16 + m) * DH + 16 + k0];
        t1[9 + 2*j] = Kt[(16 + m) * DH + 17 + k0];
      }
      s0 = wmma_bf16(qa, __builtin_bit_cast(v16bf, t0), s0);
      s1 = wmma_bf16(qa, __builtin_bit_cast(v16bf, t1), s1);
    }

    // mask + scale + online softmax; write P straight to LDS (no P arrays)
    const int ncol0 = kb + m;
    const int ncol1 = ncol0 + 16;
    unsigned short* Pw = &Ps[wave][0];
    v8f alphav;
#pragma unroll
    for (int r = 0; r < 8; ++r) {
      int qr = qrow_base + r;
      float v0 = (ncol0 <= qr) ? s0[r] * ATTN_SCALE : -__builtin_inff();
      float v1 = (ncol1 <= qr) ? s1[r] * ATTN_SCALE : -__builtin_inff();
      float mx = fmaxf(v0, v1);
#pragma unroll
      for (int dd = 8; dd >= 1; dd >>= 1) mx = fmaxf(mx, __shfl_xor(mx, dd, 16));
      float mnew  = fmaxf(mstat[r], mx);
      float alpha = __expf(fminf(mstat[r] - mnew, 0.0f));  // -inf-(-inf)->nan->min->0->exp->1
      float e0 = (v0 > -__builtin_inff()) ? __expf(v0 - mnew) : 0.0f;
      float e1 = (v1 > -__builtin_inff()) ? __expf(v1 - mnew) : 0.0f;
      float rs = e0 + e1;
#pragma unroll
      for (int dd = 8; dd >= 1; dd >>= 1) rs += __shfl_xor(rs, dd, 16);
      lstat[r] = lstat[r] * alpha + rs;
      mstat[r] = mnew;
      alphav[r] = alpha;
      int rl = (hh ? 8 : 0) + r;
      Pw[rl * 32 + m]      = f2bf_u(e0);
      Pw[rl * 32 + 16 + m] = f2bf_u(e1);
    }
#pragma unroll
    for (int f = 0; f < 8; ++f) o[f] *= alphav;
    __syncthreads();

    // O += P * V
    v16bf pf = load_frag_a(Pw, 32);
#pragma unroll
    for (int f = 0; f < 8; ++f) {
      v16u tv;
#pragma unroll
      for (int j = 0; j < 4; ++j) {
        int k0  = hh * 8 + 2 * j;
        int col = f * 16 + m;
        tv[2*j]     = Vt[(k0)      * DH + col];
        tv[2*j + 1] = Vt[(k0 + 1)  * DH + col];
        tv[8 + 2*j] = Vt[(k0 + 16) * DH + col];
        tv[9 + 2*j] = Vt[(k0 + 17) * DH + col];
      }
      o[f] = wmma_bf16(pf, __builtin_bit_cast(v16bf, tv), o[f]);
    }
    __syncthreads();  // all reads of buffer `cur` done before it is refilled
  }

  // normalize + store attn in (b, t, h*128+d) layout for output GEMM
#pragma unroll
  for (int r = 0; r < 8; ++r) {
    float invl = (lstat[r] > 0.0f) ? 1.0f / lstat[r] : 0.0f;
    int trow = qrow_base + r;
    size_t base = ((size_t)b * TT + trow) * (NQ * DH) + (size_t)h * DH;
#pragma unroll
    for (int f = 0; f < 8; ++f)
      attn_out[base + f * 16 + m] = f2bf_u(o[f][r] * invl);
  }
}

// ---------------- launch ----------------
extern "C" void kernel_launch(void* const* d_in, const int* in_sizes, int n_in,
                              void* d_out, int out_size, void* d_ws, size_t ws_size,
                              hipStream_t stream) {
  const float* x   = (const float*)d_in[0];
  const float* Wq  = (const float*)d_in[1];
  const float* Wkv = (const float*)d_in[2];
  const float* Wo  = (const float*)d_in[3];
  const float* qw  = (const float*)d_in[4];
  const float* kw  = (const float*)d_in[5];
  float* out = (float*)d_out;

  // workspace layout (bytes):
  char* ws = (char*)d_ws;
  float*          qproj  = (float*)(ws + 0);                  // 4096*2048*4  = 33554432
  float*          kvproj = (float*)(ws + 33554432);           // 4096*1024*4  = 16777216
  unsigned short* qn     = (unsigned short*)(ws + 50331648);  // 2*16*2048*128*2 = 16777216
  unsigned short* kn     = (unsigned short*)(ws + 67108864);  // 4194304
  unsigned short* vn     = (unsigned short*)(ws + 71303168);  // 4194304
  unsigned short* attn   = (unsigned short*)(ws + 75497472);  // 16777216 (end 92274688)

  // q = x @ Wq ; kv = x @ Wkv
  dim3 gq(MROWS / 64, (NQ * DH) / 64);
  gemm_wmma<false><<<gq, 256, 0, stream>>>(x, Wq, qproj, MROWS, NQ * DH, DM);
  dim3 gkv(MROWS / 64, (2 * NKV * DH) / 64);
  gemm_wmma<false><<<gkv, 256, 0, stream>>>(x, Wkv, kvproj, MROWS, 2 * NKV * DH, DM);

  // RMSNorm + RoPE + bf16 pack
  normrope<<<BB * TT * 24, DH, 0, stream>>>(qproj, kvproj, qw, kw, qn, kn, vn);

  // causal GQA flash attention
  dim3 ga(TT / 128, NQ, BB);
  attn_wmma<<<ga, 256, 0, stream>>>(qn, kn, vn, attn);

  // out = attn @ Wo
  dim3 go(MROWS / 64, DM / 64);
  gemm_wmma<true><<<go, 256, 0, stream>>>(attn, Wo, out, MROWS, DM, DM);
}
